// SpaTrans_39109972197566
// MI455X (gfx1250) — matrix operand order repaired and checked by reading, not compiled, not run on hardware
//
#include <hip/hip_runtime.h>
#include <hip/hip_bf16.h>

typedef __attribute__((ext_vector_type(16))) _Float16 v16h;
typedef __attribute__((ext_vector_type(8)))  _Float16 v8h;
typedef __attribute__((ext_vector_type(4)))  _Float16 v4h;
typedef __attribute__((ext_vector_type(8)))  float    v8f;

// ---------------------------------------------------------------------------
// wave32 reduction helpers
// ---------------------------------------------------------------------------
__device__ __forceinline__ float wave_sum(float v) {
#pragma unroll
  for (int o = 16; o > 0; o >>= 1) v += __shfl_xor(v, o, 32);
  return v;
}
__device__ __forceinline__ float wave_max(float v) {
#pragma unroll
  for (int o = 16; o > 0; o >>= 1) v = fmaxf(v, __shfl_xor(v, o, 32));
  return v;
}

// ---------------------------------------------------------------------------
// generic f32 -> f16 convert
// ---------------------------------------------------------------------------
__global__ void cvt_f16_kernel(const float* __restrict__ in, _Float16* __restrict__ out,
                               long long count) {
  long long i = (long long)blockIdx.x * 256 + threadIdx.x;
  if (i < count) out[i] = (_Float16)in[i];
}

// ---------------------------------------------------------------------------
// unfold 3x3 (torch F.unfold layout: channel-slowest, kernel-fastest)
// buffer [1,64,25,32,32] -> Upat f16 [25][576][1024]
// ---------------------------------------------------------------------------
__global__ void unfold_kernel(const float* __restrict__ buf, _Float16* __restrict__ up) {
  long long idx = (long long)blockIdx.x * 256 + threadIdx.x;   // 25*576*1024
  int l  = (int)(idx & 1023);
  long long t = idx >> 10;
  int k  = (int)(t % 576);
  int n  = (int)(t / 576);
  int c  = k / 9, r = k % 9;
  int ki = r / 3, kj = r % 3;
  int h  = l >> 5, w = l & 31;
  int hh = h - 1 + ki, ww = w - 1 + kj;
  float v = 0.f;
  if (hh >= 0 && hh < 32 && ww >= 0 && ww < 32)
    v = buf[((long long)(c * 25 + n)) * 1024 + hh * 32 + ww];
  up[idx] = (_Float16)v;
}

// ---------------------------------------------------------------------------
// Generic batched WMMA GEMM:  C[b][m][col] (+)= sum_k A[b][m][k] * B[b][k][col]
//
// Block: 128 threads (4 waves); tile 64(M) x 128(N).
// LDS tiles stored PRE-SWIZZLED in fragment order (each fragment-lane's 16
// halves contiguous -> fragment load = 2x ds_load_b128).
//   A frag (16x32 f16): lane=(m&15)+16*hf, e -> k = 8*hf + 16*(e>=8) + (e&7)
//   B frag (32x16 f16): lane=(n&15)+16*hf, e -> k = 8*hf + 16*(e>=8) + (e&7)
// Staging is guard-free in the main K loop (out-of-range M rows / N cols only
// feed C elements the store guard drops; addresses are clamped). K-edge
// (K % 32 != 0) handled by one branchless guarded epilogue step.
// Layout-specialized vectorized staging:
//   A: csA==1 -> 4x v4h per fragment-lane; rsA==1 -> cooperative v8h row loads
//   B: rsB==1 -> 2x v8h per fragment-lane; csB==1 -> cooperative v8h row loads
// ---------------------------------------------------------------------------
__global__ void gemm_wmma_kernel(const _Float16* __restrict__ A,
                                 const _Float16* __restrict__ B,
                                 float* __restrict__ C,
                                 int M, int Ncols, int K,
                                 long long rsA, long long csA, long long sAb,
                                 long long rsB, long long csB, long long sBb,
                                 long long ldc, long long sCb, int accumulate) {
  __shared__ __align__(32) _Float16 As[4 * 32 * 16];   // 4 KB
  __shared__ __align__(32) _Float16 Bs[8 * 32 * 16];   // 8 KB

  const int tid    = threadIdx.x;
  const int wave   = tid >> 5;
  const int lane   = tid & 31;
  const int ml     = lane & 15;
  const int hf     = lane >> 4;
  const int blockM = blockIdx.y * 64;
  const int blockN = blockIdx.x * 128;
  const long long batch = blockIdx.z;

  const _Float16* Ab = A + batch * sAb;
  const _Float16* Bb = B + batch * sBb;
  float*          Cb = C + batch * sCb;

  v8f acc[8];
#pragma unroll
  for (int nb = 0; nb < 8; ++nb)
#pragma unroll
    for (int r = 0; r < 8; ++r) acc[nb][r] = 0.f;

  auto compute = [&]() {
    v16h afr = *(const v16h*)(As + (wave * 32 + lane) * 16);
#pragma unroll
    for (int nb = 0; nb < 8; ++nb) {
      v16h bfr = *(const v16h*)(Bs + (nb * 32 + lane) * 16);
      acc[nb] = __builtin_amdgcn_wmma_f32_16x16x32_f16(
          false, afr, false, bfr, (short)0, acc[nb], false, false);
    }
  };

  // precomputed per-thread staging coordinates
  const int sA   = tid >> 5;                 // A dest M-subtile (fast path)
  const int mA   = blockM + sA * 16 + ml;    // A dest row
  const long long mAc = (mA < M) ? mA : (M - 1);   // clamped (rows >=M unused)
  const int khA  = hf * 8;
  const int klc  = tid >> 2;                 // cooperative: k-row 0..31
  const int moc  = (tid & 3) * 16;           // cooperative A: m offset
  const int noc  = (tid & 3) * 32;           // cooperative B: n offset

  const int Kmain = K & ~31;

  for (int kk = 0; kk < Kmain; kk += 32) {
    // ---------------- stage A ----------------
    if (csA == 1) {
      const _Float16* ap = Ab + mAc * rsA + (kk + khA);
      v4h a0 = *(const v4h*)(ap);
      v4h a1 = *(const v4h*)(ap + 4);
      v4h a2 = *(const v4h*)(ap + 16);
      v4h a3 = *(const v4h*)(ap + 20);
      _Float16* d = As + tid * 16;
      *(v4h*)(d)      = a0;
      *(v4h*)(d + 4)  = a1;
      *(v4h*)(d + 8)  = a2;
      *(v4h*)(d + 12) = a3;
    } else {  // rsA == 1: cooperative, coalesced rows of A^T
      int mb = blockM + moc;
      if (mb + 16 > M) mb = (M > 16) ? (M - 16) : 0;   // address clamp only
      const _Float16* ap = Ab + (long long)(kk + klc) * csA + mb;
      v8h r0 = *(const v8h*)(ap);
      v8h r1 = *(const v8h*)(ap + 8);
      int hfA = (klc >> 3) & 1;
      int eA  = ((klc >> 4) << 3) + (klc & 7);
#pragma unroll
      for (int i = 0; i < 16; ++i) {
        int m = moc + i;
        _Float16 v = (i < 8) ? r0[i] : r1[i - 8];
        As[(((m >> 4) * 32) + (m & 15) + 16 * hfA) * 16 + eA] = v;
      }
    }
    // ---------------- stage B ----------------
    if (rsB == 1) {  // transposed B: fragment-lane is contiguous in memory
#pragma unroll
      for (int p = 0; p < 2; ++p) {
        int idx = tid + p * 128;
        int nb  = idx >> 5;
        int ln  = idx & 31;
        int n   = blockN + nb * 16 + (ln & 15);
        if (n >= Ncols) n = Ncols - 1;                 // address clamp only
        const _Float16* bp = Bb + (long long)n * csB + kk + (ln >> 4) * 8;
        v8h b0 = *(const v8h*)(bp);
        v8h b1 = *(const v8h*)(bp + 16);
        _Float16* d = Bs + idx * 16;
        *(v8h*)(d)     = b0;
        *(v8h*)(d + 8) = b1;
      }
    } else {  // csB == 1: cooperative, coalesced B rows
      int nb0 = blockN + noc;
      if (nb0 + 32 > Ncols) nb0 = (Ncols > 32) ? (Ncols - 32) : 0;
      const _Float16* bp = Bb + (long long)(kk + klc) * rsB + nb0;
      v8h q0 = *(const v8h*)(bp);
      v8h q1 = *(const v8h*)(bp + 8);
      v8h q2 = *(const v8h*)(bp + 16);
      v8h q3 = *(const v8h*)(bp + 24);
      int hfB = (klc >> 3) & 1;
      int eB  = ((klc >> 4) << 3) + (klc & 7);
#pragma unroll
      for (int i = 0; i < 32; ++i) {
        int n = noc + i;
        _Float16 v = (i < 8) ? q0[i] : (i < 16) ? q1[i - 8] : (i < 24) ? q2[i - 16] : q3[i - 24];
        Bs[(((n >> 4) * 32) + (n & 15) + 16 * hfB) * 16 + eB] = v;
      }
    }
    __syncthreads();
    compute();
    __syncthreads();
  }

  // ---------------- K-edge epilogue (K % 32 != 0): branchless guarded gather --
  if (Kmain < K) {
    {
      v16h tmp;
#pragma unroll
      for (int e = 0; e < 16; ++e) {
        int k = Kmain + khA + ((e >> 3) << 4) + (e & 7);
        long long kc = (k < K) ? k : (K - 1);
        _Float16 v = Ab[mAc * rsA + kc * csA];
        tmp[e] = (k < K) ? v : (_Float16)0;
      }
      *(v16h*)(As + tid * 16) = tmp;
    }
#pragma unroll
    for (int p = 0; p < 2; ++p) {
      int idx = tid + p * 128;
      int nb  = idx >> 5;
      int ln  = idx & 31;
      int n   = blockN + nb * 16 + (ln & 15);
      long long nc = (n < Ncols) ? n : (Ncols - 1);
      int kh2 = (ln >> 4) * 8;
      v16h tmp;
#pragma unroll
      for (int e = 0; e < 16; ++e) {
        int k = Kmain + kh2 + ((e >> 3) << 4) + (e & 7);
        long long kc = (k < K) ? k : (K - 1);
        _Float16 v = Bb[kc * rsB + nc * csB];
        tmp[e] = (k < K) ? v : (_Float16)0;
      }
      *(v16h*)(Bs + idx * 16) = tmp;
    }
    __syncthreads();
    compute();
    __syncthreads();
  }

  // ---- store: C layout lanes 0-15 -> M=vgpr, lanes 16-31 -> M=vgpr+8 ----
#pragma unroll
  for (int nb = 0; nb < 8; ++nb)
#pragma unroll
    for (int r = 0; r < 8; ++r) {
      int m   = blockM + wave * 16 + r + (hf ? 8 : 0);
      int col = blockN + nb * 16 + ml;
      if (m < M && col < Ncols) {
        long long idx = (long long)m * ldc + col;
        float v = acc[nb][r];
        if (accumulate) Cb[idx] += v; else Cb[idx] = v;
      }
    }
}

// ---------------------------------------------------------------------------
// LayerNorm over D=128 per (n,l); tok f32 [25][128][1024] -> f16 same layout
// ---------------------------------------------------------------------------
__global__ void layernorm_f16_kernel(const float* __restrict__ tok,
                                     const float* __restrict__ w,
                                     const float* __restrict__ b,
                                     _Float16* __restrict__ out) {
  int wid  = blockIdx.x * 8 + (threadIdx.x >> 5);   // 25*1024 waves
  int lane = threadIdx.x & 31;
  int n = wid >> 10, l = wid & 1023;
  const float* base = tok + (long long)n * 131072 + l;
  float v[4]; float s = 0.f;
#pragma unroll
  for (int i = 0; i < 4; ++i) { v[i] = base[(lane + 32 * i) * 1024]; s += v[i]; }
  s = wave_sum(s);
  float mean = s * (1.f / 128.f);
  float q = 0.f;
#pragma unroll
  for (int i = 0; i < 4; ++i) { float d = v[i] - mean; q += d * d; }
  q = wave_sum(q);
  float rstd = rsqrtf(q * (1.f / 128.f) + 1e-5f);
  _Float16* ob = out + (long long)n * 131072 + l;
#pragma unroll
  for (int i = 0; i < 4; ++i) {
    int c = lane + 32 * i;
    ob[c * 1024] = (_Float16)((v[i] - mean) * rstd * w[c] + b[c]);
  }
}

// ---------------------------------------------------------------------------
// depthwise 3x3, SAME padding; in/out f32 [25][CH][1024], w [CH][9]
// ---------------------------------------------------------------------------
__global__ void dwconv3x3_kernel(const float* __restrict__ in,
                                 const float* __restrict__ w,
                                 float* __restrict__ out, int CH) {
  long long idx = (long long)blockIdx.x * 256 + threadIdx.x;  // 25*CH*1024
  int l = (int)(idx & 1023);
  long long t = idx >> 10;
  int ch = (int)(t % CH);
  int n  = (int)(t / CH);
  int h = l >> 5, ww = l & 31;
  const float* ib = in + ((long long)n * CH + ch) * 1024;
  const float* wk = w + (long long)ch * 9;
  float s = 0.f;
#pragma unroll
  for (int ki = 0; ki < 3; ++ki) {
    int hh = h - 1 + ki;
    if (hh < 0 || hh >= 32) continue;
#pragma unroll
    for (int kj = 0; kj < 3; ++kj) {
      int wj = ww - 1 + kj;
      if (wj < 0 || wj >= 32) continue;
      s += ib[hh * 32 + wj] * wk[ki * 3 + kj];
    }
  }
  out[idx] = s;
}

// ---------------------------------------------------------------------------
// split qkv into per-head Q,K,V; l2-normalize q,k over c (32); write f16
// ---------------------------------------------------------------------------
__global__ void qkv_head_norm_kernel(const float* __restrict__ qkvd,
                                     _Float16* __restrict__ Qn,
                                     _Float16* __restrict__ Kn,
                                     _Float16* __restrict__ Vn) {
  int wid  = blockIdx.x * 8 + (threadIdx.x >> 5);   // 25*4*1024 waves
  int lane = threadIdx.x & 31;
  int l = wid & 1023;
  int h = (wid >> 10) & 3;
  int n = wid >> 12;
  long long qoff = ((long long)(n * 384 + h * 32 + lane)) * 1024 + l;
  float qv = qkvd[qoff];
  float kv = qkvd[qoff + 131072];
  float vv = qkvd[qoff + 262144];
  float dq = fmaxf(sqrtf(wave_sum(qv * qv)), 1e-12f);
  float dk = fmaxf(sqrtf(wave_sum(kv * kv)), 1e-12f);
  long long o = ((long long)((n * 4 + h) * 32 + lane)) * 1024 + l;
  Qn[o] = (_Float16)(qv / dq);
  Kn[o] = (_Float16)(kv / dk);
  Vn[o] = (_Float16)vv;
}

// ---------------------------------------------------------------------------
// softmax over last dim of S f32 [4][1024][1024] with per-head temperature
// ---------------------------------------------------------------------------
__global__ void softmax_rows_kernel(const float* __restrict__ S,
                                    _Float16* __restrict__ P,
                                    const float* __restrict__ temp) {
  __shared__ float red[8];
  int h = blockIdx.x >> 10;
  int l = blockIdx.x & 1023;
  const float* row  = S + ((long long)h * 1024 + l) * 1024;
  _Float16*    prow = P + ((long long)h * 1024 + l) * 1024;
  float t = temp[h];
  int tid = threadIdx.x;
  float vals[4];
  float mx = -3.4e38f;
#pragma unroll
  for (int i = 0; i < 4; ++i) { vals[i] = t * row[tid + 256 * i]; mx = fmaxf(mx, vals[i]); }
  mx = wave_max(mx);
  if ((tid & 31) == 0) red[tid >> 5] = mx;
  __syncthreads();
  float bmax = red[0];
#pragma unroll
  for (int i = 1; i < 8; ++i) bmax = fmaxf(bmax, red[i]);
  __syncthreads();
  float s = 0.f;
#pragma unroll
  for (int i = 0; i < 4; ++i) { vals[i] = __expf(vals[i] - bmax); s += vals[i]; }
  s = wave_sum(s);
  if ((tid & 31) == 0) red[tid >> 5] = s;
  __syncthreads();
  float tot = 0.f;
#pragma unroll
  for (int i = 0; i < 8; ++i) tot += red[i];
  float inv = 1.f / tot;
#pragma unroll
  for (int i = 0; i < 4; ++i) prow[tid + 256 * i] = (_Float16)(vals[i] * inv);
}

// ---------------------------------------------------------------------------
// GEGLU: g = gelu_exact(y1) * y2
// ---------------------------------------------------------------------------
__global__ void geglu_f16_kernel(const float* __restrict__ y, _Float16* __restrict__ g) {
  long long idx = (long long)blockIdx.x * 256 + threadIdx.x;  // 25*340*1024
  int l = (int)(idx & 1023);
  long long t = idx >> 10;
  int j = (int)(t % 340);
  int n = (int)(t / 340);
  long long b1 = ((long long)(n * 680 + j)) * 1024 + l;
  float x1 = y[b1];
  float x2 = y[b1 + 340 * 1024];
  float ge = 0.5f * x1 * (1.f + erff(x1 * 0.70710678118654752f));
  g[((long long)(n * 340 + j)) * 1024 + l] = (_Float16)(ge * x2);
}

// ---------------------------------------------------------------------------
// host-side launch
// ---------------------------------------------------------------------------
static inline void gemm(hipStream_t s, const _Float16* A, const _Float16* B, float* C,
                        int M, int N, int K,
                        long long rsA, long long csA, long long sAb,
                        long long rsB, long long csB, long long sBb,
                        long long ldc, long long sCb, int batches, int accum) {
  dim3 g((N + 127) / 128, (M + 63) / 64, batches);
  gemm_wmma_kernel<<<g, dim3(128), 0, s>>>(A, B, C, M, N, K,
                                           rsA, csA, sAb, rsB, csB, sBb, ldc, sCb, accum);
}

static inline void cvt(hipStream_t s, const float* in, _Float16* out, long long n) {
  cvt_f16_kernel<<<dim3((unsigned)((n + 255) / 256)), dim3(256), 0, s>>>(in, out, n);
}

extern "C" void kernel_launch(void* const* d_in, const int* in_sizes, int n_in,
                              void* d_out, int out_size, void* d_ws, size_t ws_size,
                              hipStream_t stream) {
  (void)in_sizes; (void)n_in; (void)out_size; (void)ws_size;

  const float* buffer      = (const float*)d_in[0];
  const float* mlp_w       = (const float*)d_in[1];
  const float* ln_attn_w   = (const float*)d_in[2];
  const float* ln_attn_b   = (const float*)d_in[3];
  const float* ln_ffn_w    = (const float*)d_in[4];
  const float* ln_ffn_b    = (const float*)d_in[5];
  const float* qkv_w       = (const float*)d_in[6];
  const float* qkv_dw_w    = (const float*)d_in[7];
  const float* temperature = (const float*)d_in[8];
  const float* proj_attn_w = (const float*)d_in[9];
  const float* ffn_in_w    = (const float*)d_in[10];
  const float* ffn_dw_w    = (const float*)d_in[11];
  const float* ffn_out_w   = (const float*)d_in[12];
  const float* proj3d_w    = (const float*)d_in[13];
  float* out = (float*)d_out;

  // ---- workspace layout (bytes, all sizes multiples of 256) ----
  char* ws = (char*)d_ws;
  size_t off = 0;
  auto alloc = [&](size_t bytes) { char* p = ws + off; off += bytes; return p; };

  float*    tok    = (float*)   alloc(25ll * 128 * 1024 * 4);   // [25][128][1024]
  _Float16* xln    = (_Float16*)alloc(25ll * 128 * 1024 * 2);   // LN output / later tok_h
  char*     r1     =            alloc(25ll * 680 * 1024 * 4);   // Upat then ffnd
  char*     r2     =            alloc(25ll * 384 * 1024 * 4);   // qkvbuf, attno(+h), gbuf
  char*     r3     =            alloc(25ll * 680 * 1024 * 4);   // qkvd, S+P, ffnbuf
  _Float16* Qn     = (_Float16*)alloc(25ll * 128 * 1024 * 2);
  _Float16* Kn     = (_Float16*)alloc(25ll * 128 * 1024 * 2);
  _Float16* Vn     = (_Float16*)alloc(25ll * 128 * 1024 * 2);
  _Float16* mlp_h  = (_Float16*)alloc(128ll * 576 * 2);
  _Float16* qkv_h  = (_Float16*)alloc(384ll * 128 * 2);
  _Float16* prj_h  = (_Float16*)alloc(128ll * 128 * 2);
  _Float16* fin_h  = (_Float16*)alloc(680ll * 128 * 2);
  _Float16* fout_h = (_Float16*)alloc(128ll * 340 * 2);
  _Float16* p3d_h  = (_Float16*)alloc(64ll * 128 * 2);

  _Float16* Upat   = (_Float16*)r1;
  float*    ffnd   = (float*)r1;
  float*    qkvbuf = (float*)r2;
  float*    attno  = (float*)r2;
  _Float16* attno_h= (_Float16*)(r2 + 25ll * 128 * 1024 * 4);
  _Float16* gbuf   = (_Float16*)r2;
  float*    qkvd   = (float*)r3;
  float*    Sbuf   = (float*)r3;
  _Float16* Pbuf   = (_Float16*)(r3 + 4ll * 1024 * 1024 * 4);
  float*    ffnbuf = (float*)r3;
  _Float16* tok_h  = xln;

  // ---- weight conversions ----
  cvt(stream, mlp_w,       mlp_h,  128ll * 576);
  cvt(stream, qkv_w,       qkv_h,  384ll * 128);
  cvt(stream, proj_attn_w, prj_h,  128ll * 128);
  cvt(stream, ffn_in_w,    fin_h,  680ll * 128);
  cvt(stream, ffn_out_w,   fout_h, 128ll * 340);
  cvt(stream, proj3d_w,    p3d_h,  64ll * 128);

  // ---- SAI2Token: unfold + token MLP ----
  unfold_kernel<<<dim3(57600), dim3(256), 0, stream>>>(buffer, Upat);
  gemm(stream, mlp_h, Upat, tok, 128, 1024, 576,
       576, 1, 0, 1024, 1, 576ll * 1024, 1024, 131072, 25, 0);

  // ---- attention branch ----
  layernorm_f16_kernel<<<dim3(3200), dim3(256), 0, stream>>>(tok, ln_attn_w, ln_attn_b, xln);
  gemm(stream, qkv_h, xln, qkvbuf, 384, 1024, 128,
       128, 1, 0, 1024, 1, 131072, 1024, 393216, 25, 0);
  dwconv3x3_kernel<<<dim3(38400), dim3(256), 0, stream>>>(qkvbuf, qkv_dw_w, qkvd, 384);
  qkv_head_norm_kernel<<<dim3(12800), dim3(256), 0, stream>>>(qkvd, Qn, Kn, Vn);

  for (int n = 0; n < 25; ++n) {
    // S[h][l][m] = sum_c Qn[l,c] * Kn[m,c]  (A transposed: rsA=1, csA=1024)
    gemm(stream, Qn + (long long)n * 131072, Kn + (long long)n * 131072, Sbuf,
         1024, 1024, 32,
         1, 1024, 32768, 1024, 1, 32768, 1024, 1048576, 4, 0);
    softmax_rows_kernel<<<dim3(4096), dim3(256), 0, stream>>>(Sbuf, Pbuf, temperature);
    // attno[n][h*32+c][l] = sum_m Vn[c,m] * P[l,m]  (B transposed: rsB=1, csB=1024)
    gemm(stream, Vn + (long long)n * 131072, Pbuf, attno + (long long)n * 131072,
         32, 1024, 1024,
         1024, 1, 32768, 1, 1024, 1048576, 1024, 32768, 4, 0);
  }
  cvt(stream, attno, attno_h, 25ll * 128 * 1024);
  gemm(stream, prj_h, attno_h, tok, 128, 1024, 128,
       128, 1, 0, 1024, 1, 131072, 1024, 131072, 25, 1);

  // ---- GDFN branch ----
  layernorm_f16_kernel<<<dim3(3200), dim3(256), 0, stream>>>(tok, ln_ffn_w, ln_ffn_b, xln);
  gemm(stream, fin_h, xln, ffnbuf, 680, 1024, 128,
       128, 1, 0, 1024, 1, 131072, 1024, 696320, 25, 0);
  dwconv3x3_kernel<<<dim3(68000), dim3(256), 0, stream>>>(ffnbuf, ffn_dw_w, ffnd, 680);
  geglu_f16_kernel<<<dim3(34000), dim3(256), 0, stream>>>(ffnd, gbuf);
  gemm(stream, fout_h, gbuf, tok, 128, 1024, 340,
       340, 1, 0, 1024, 1, 348160, 1024, 131072, 25, 1);

  // ---- Token2SAI + proj3d ----
  cvt(stream, tok, tok_h, 25ll * 128 * 1024);
  gemm(stream, p3d_h, tok_h, out, 64, 1024, 128,
       128, 1, 0, 1024, 1, 131072, 25600, 1024, 25, 0);
}
